// Model_51436528337205
// MI455X (gfx1250) — compile-verified, matrix operand order checked by reading, not run on hardware
//
#include <hip/hip_runtime.h>

// ---------------------------------------------------------------------------
// MHA block for gfx1250 (MI455X): f16 WMMA (v_wmma_f32_16x16x32_f16), flash
// attention (no materialized S x S matrix), wave32, no LDS needed thanks to
// the transposed-score trick (S^T C-layout == P^T B-fragment layout).
// Round 2: flash processes 32 queries/wave (1 b128 load per WMMA, half the
// K/V traffic) + global_prefetch of upcoming K/V blocks.
// ---------------------------------------------------------------------------

typedef __attribute__((ext_vector_type(16))) _Float16 v16h;
typedef __attribute__((ext_vector_type(8)))  _Float16 v8h;
typedef __attribute__((ext_vector_type(8)))  float    v8f;

#define SEQ  4096
#define HID  768
#define NH   12
#define HD   64

union FragU { v16h v; v8h h[2]; };

// Fragment loader for A- and B-fragments of v_wmma_f32_16x16x32_f16.
// Both have the same per-lane structure: lane holds row (A) / column (B)
// index = lane%16; K split in 4 chunks of 8: lanes 0-15 get K chunks {0,2},
// lanes 16-31 get {1,3}. Each chunk is 8 contiguous f16 = one 16B load.
__device__ __forceinline__ v16h load_frag(const _Float16* __restrict__ rowp,
                                          int k0, int lane) {
  const int kb = (lane & 16) ? 8 : 0;
  FragU f;
  f.h[0] = *(const v8h*)(rowp + k0 + kb);
  f.h[1] = *(const v8h*)(rowp + k0 + 16 + kb);
  return f.v;
}

__device__ __forceinline__ v8f wmma16(v16h a, v16h b, v8f c) {
  return __builtin_amdgcn_wmma_f32_16x16x32_f16(
      /*neg_a=*/false, a, /*neg_b=*/false, b,
      /*c_mod=*/(short)0, c, /*reuse_a=*/false, /*reuse_b=*/false);
}

// ---------------------------------------------------------------------------
// Prep kernels: f32 -> f16 convert / transpose (tiny; bandwidth irrelevant)
// ---------------------------------------------------------------------------
__global__ void cvt_f16_kernel(const float* __restrict__ in,
                               _Float16* __restrict__ out, int n) {
  int i = blockIdx.x * 256 + threadIdx.x;
  if (i < n) out[i] = (_Float16)in[i];
}

// out[N][K] = in[K][N]  (i indexes out flat: n = i/K, k = i%K)
__global__ void transpose_f16_kernel(const float* __restrict__ in,
                                     _Float16* __restrict__ out, int K, int N) {
  int i = blockIdx.x * 256 + threadIdx.x;
  if (i < N * K) {
    int n = i / K, k = i % K;
    out[i] = (_Float16)in[k * N + n];
  }
}

// ---------------------------------------------------------------------------
// QKV GEMM: C[4096,2304] = xh[4096,768] @ wqkvT[2304,768]^T + b_qkv
// Wave computes 32(M) x 64(N). Epilogue scatters into per-head Q (x 1/8),
// K (row-major [h][S][64]) and V transposed ([h][64][S]), all f16.
// ---------------------------------------------------------------------------
__global__ __launch_bounds__(256)
void gemm_qkv_kernel(const _Float16* __restrict__ xh,
                     const _Float16* __restrict__ wT,
                     const float* __restrict__ bias,
                     _Float16* __restrict__ Qh,
                     _Float16* __restrict__ Kh,
                     _Float16* __restrict__ Vt) {
  const int lane = threadIdx.x & 31;
  const int gw   = blockIdx.x * 8 + (threadIdx.x >> 5);
  const int Mt   = SEQ / 32;            // 128
  const int mt   = gw % Mt;             // M tile (32 rows)
  const int nt   = gw / Mt;             // N tile (64 cols), 0..35
  const int lm   = lane & 15;

  const _Float16* a0p = xh + (size_t)(mt * 32 + lm) * HID;
  const _Float16* a1p = a0p + 16 * HID;
  const _Float16* bp[4];
#pragma unroll
  for (int j = 0; j < 4; ++j)
    bp[j] = wT + (size_t)(nt * 64 + j * 16 + lm) * HID;

  v8f acc[2][4];
#pragma unroll
  for (int i = 0; i < 2; ++i)
#pragma unroll
    for (int j = 0; j < 4; ++j) acc[i][j] = (v8f)0.0f;

  for (int k0 = 0; k0 < HID; k0 += 32) {
    v16h a0 = load_frag(a0p, k0, lane);
    v16h a1 = load_frag(a1p, k0, lane);
#pragma unroll
    for (int j = 0; j < 4; ++j) {
      v16h b = load_frag(bp[j], k0, lane);
      acc[0][j] = wmma16(a0, b, acc[0][j]);
      acc[1][j] = wmma16(a1, b, acc[1][j]);
    }
  }

  // Epilogue: col = nt*64 + j*16 + lm ; sec = col/768 ; head = (col%768)/64
  const int sec  = nt / 12;
  const int head = nt % 12;
  const int mo   = (lane & 16) ? 8 : 0;
#pragma unroll
  for (int i = 0; i < 2; ++i) {
#pragma unroll
    for (int j = 0; j < 4; ++j) {
      const int dd = j * 16 + lm;                 // dim within head, 0..63
      const float bv = bias[nt * 64 + dd];
#pragma unroll
      for (int d = 0; d < 8; ++d) {
        const int row = mt * 32 + i * 16 + d + mo;  // sequence position
        const float v = acc[i][j][d] + bv;
        if (sec == 0) {
          Qh[((size_t)head * SEQ + row) * HD + dd] = (_Float16)(v * 0.125f);
        } else if (sec == 1) {
          Kh[((size_t)head * SEQ + row) * HD + dd] = (_Float16)v;
        } else {
          Vt[((size_t)head * HD + dd) * SEQ + row] = (_Float16)v;
        }
      }
    }
  }
}

// ---------------------------------------------------------------------------
// Flash attention: 1 wave per (head, 32-query tile); kv blocks of 32 keys.
//   S^T = K_blk @ Q^T  (per query group: 2 C-tiles, 4 WMMAs) -> online
//   softmax -> pack P^T B-fragment (free: S^T C-layout == P^T B-frag layout)
//   -> O^T += V^T_blk @ P^T (4 WMMAs per group, V fragments shared).
// Per kv block: 16 WMMAs, 16 b128 fragment loads (1 load/WMMA).
// alpha/lsum are uniform per lane (lane%16 = query), so no broadcasts.
// ---------------------------------------------------------------------------
__global__ __launch_bounds__(256)
void flash_attn_kernel(const _Float16* __restrict__ Qh,
                       const _Float16* __restrict__ Kh,
                       const _Float16* __restrict__ Vt,
                       _Float16* __restrict__ Oh) {
  const int lane = threadIdx.x & 31;
  const int gw   = blockIdx.x * 8 + (threadIdx.x >> 5);
  const int head = gw >> 7;            // 128 query tiles (of 32) per head
  const int qt   = gw & 127;
  const int lm   = lane & 15;

  const _Float16* Kp = Kh + (size_t)head * SEQ * HD;
  const _Float16* Vp = Vt + (size_t)head * HD * SEQ;

  // Q^T B-fragments (column = query) for 2 query groups x dims {0-31, 32-63}
  v16h qf[2][2];
#pragma unroll
  for (int g = 0; g < 2; ++g) {
    const _Float16* Qp =
        Qh + (size_t)head * SEQ * HD + (size_t)(qt * 32 + g * 16 + lm) * HD;
    qf[g][0] = load_frag(Qp, 0, lane);
    qf[g][1] = load_frag(Qp, 32, lane);
  }

  v8f oacc[2][4];
#pragma unroll
  for (int g = 0; g < 2; ++g)
#pragma unroll
    for (int t = 0; t < 4; ++t) oacc[g][t] = (v8f)0.0f;
  float mrun[2] = {-__builtin_inff(), -__builtin_inff()};
  float lsum[2] = {0.0f, 0.0f};

  for (int kb = 0; kb < SEQ; kb += 32) {
    // Prefetch K/V 4 kv-blocks ahead (wrapped: stays inside the buffers);
    // lowers to global_prefetch_b8.
    {
      const int kn = (kb + 128) & (SEQ - 1);
      __builtin_prefetch(Kp + (size_t)(kn + lm) * HD, 0, 1);
      __builtin_prefetch(Kp + (size_t)(kn + 16 + lm) * HD, 0, 1);
#pragma unroll
      for (int t = 0; t < 4; ++t)
        __builtin_prefetch(Vp + (size_t)(t * 16 + lm) * SEQ + kn, 0, 1);
    }

    // K A-fragments: [key half-block][dim k-step], shared by both q groups
    const _Float16* k0p = Kp + (size_t)(kb + lm) * HD;
    const _Float16* k1p = Kp + (size_t)(kb + 16 + lm) * HD;
    v16h ak[2][2];
    ak[0][0] = load_frag(k0p, 0, lane);  ak[0][1] = load_frag(k0p, 32, lane);
    ak[1][0] = load_frag(k1p, 0, lane);  ak[1][1] = load_frag(k1p, 32, lane);

    v16h pf[2];
#pragma unroll
    for (int g = 0; g < 2; ++g) {
      v8f z = (v8f)0.0f;
      v8f st0 = wmma16(ak[0][0], qf[g][0], z);
      st0     = wmma16(ak[0][1], qf[g][1], st0);
      v8f st1 = wmma16(ak[1][0], qf[g][0], z);
      st1     = wmma16(ak[1][1], qf[g][1], st1);

      // --- online softmax (row = query = lane%16; partner = lane^16) ---
      float mx = st0[0];
#pragma unroll
      for (int d = 1; d < 8; ++d) mx = fmaxf(mx, st0[d]);
#pragma unroll
      for (int d = 0; d < 8; ++d) mx = fmaxf(mx, st1[d]);
      mx = fmaxf(mx, __shfl_xor(mx, 16, 32));
      const float mnew  = fmaxf(mrun[g], mx);
      const float alpha = __expf(mrun[g] - mnew);

      float p[16];
      float psum = 0.0f;
#pragma unroll
      for (int d = 0; d < 8; ++d) { p[d]     = __expf(st0[d] - mnew); psum += p[d]; }
#pragma unroll
      for (int d = 0; d < 8; ++d) { p[8 + d] = __expf(st1[d] - mnew); psum += p[8 + d]; }
      psum += __shfl_xor(psum, 16, 32);
      lsum[g] = lsum[g] * alpha + psum;
      mrun[g] = mnew;

      // P^T B-fragment: exactly the S^T C-layout, just cvt f32->f16
#pragma unroll
      for (int d = 0; d < 16; ++d) pf[g][d] = (_Float16)p[d];

      // rescale this group's O^T accumulators (alpha uniform per lane)
#pragma unroll
      for (int t = 0; t < 4; ++t) oacc[g][t] = oacc[g][t] * alpha;
    }

    // O^T tiles: A = V^T rows (dims), K-dim = 32 keys; V frags shared by
    // both query groups.
#pragma unroll
    for (int t = 0; t < 4; ++t) {
      v16h av = load_frag(Vp + (size_t)(t * 16 + lm) * SEQ, kb, lane);
      oacc[0][t] = wmma16(av, pf[0], oacc[0][t]);
      oacc[1][t] = wmma16(av, pf[1], oacc[1][t]);
    }
  }

  const int mo = (lane & 16) ? 8 : 0;
#pragma unroll
  for (int g = 0; g < 2; ++g) {
    const float inv = 1.0f / lsum[g];
    _Float16* Op = Oh + (size_t)(qt * 32 + g * 16 + lm) * HID + head * HD;
#pragma unroll
    for (int t = 0; t < 4; ++t)
#pragma unroll
      for (int d = 0; d < 8; ++d)
        Op[t * 16 + d + mo] = (_Float16)(oacc[g][t][d] * inv);
  }
}

// ---------------------------------------------------------------------------
// Output projection: out[4096,768] = Oh[4096,768] @ wprojT[768,768]^T + b_proj
// ---------------------------------------------------------------------------
__global__ __launch_bounds__(256)
void gemm_proj_kernel(const _Float16* __restrict__ Oh,
                      const _Float16* __restrict__ wT,
                      const float* __restrict__ bias,
                      float* __restrict__ out) {
  const int lane = threadIdx.x & 31;
  const int gw   = blockIdx.x * 8 + (threadIdx.x >> 5);
  const int Mt   = SEQ / 32;            // 128
  const int mt   = gw % Mt;
  const int nt   = gw / Mt;             // 0..11
  const int lm   = lane & 15;

  const _Float16* a0p = Oh + (size_t)(mt * 32 + lm) * HID;
  const _Float16* a1p = a0p + 16 * HID;
  const _Float16* bp[4];
#pragma unroll
  for (int j = 0; j < 4; ++j)
    bp[j] = wT + (size_t)(nt * 64 + j * 16 + lm) * HID;

  v8f acc[2][4];
#pragma unroll
  for (int i = 0; i < 2; ++i)
#pragma unroll
    for (int j = 0; j < 4; ++j) acc[i][j] = (v8f)0.0f;

  for (int k0 = 0; k0 < HID; k0 += 32) {
    v16h a0 = load_frag(a0p, k0, lane);
    v16h a1 = load_frag(a1p, k0, lane);
#pragma unroll
    for (int j = 0; j < 4; ++j) {
      v16h b = load_frag(bp[j], k0, lane);
      acc[0][j] = wmma16(a0, b, acc[0][j]);
      acc[1][j] = wmma16(a1, b, acc[1][j]);
    }
  }

  const int mo = (lane & 16) ? 8 : 0;
#pragma unroll
  for (int i = 0; i < 2; ++i) {
#pragma unroll
    for (int j = 0; j < 4; ++j) {
      const int col = nt * 64 + j * 16 + lm;
      const float bv = bias[col];
#pragma unroll
      for (int d = 0; d < 8; ++d) {
        const int row = mt * 32 + i * 16 + d + mo;
        out[(size_t)row * HID + col] = acc[i][j][d] + bv;
      }
    }
  }
}

// ---------------------------------------------------------------------------
// Host launcher. Workspace layout (f16 elements), total ~36.2 MB:
//   xh[S*HID] | wqkvT[3*HID*HID] | wprojT[HID*HID] |
//   Qh[S*HID] | Kh[S*HID] | Vt[S*HID] | Oh[S*HID]
// ---------------------------------------------------------------------------
extern "C" void kernel_launch(void* const* d_in, const int* in_sizes, int n_in,
                              void* d_out, int out_size, void* d_ws, size_t ws_size,
                              hipStream_t stream) {
  const float* x      = (const float*)d_in[0];
  const float* w_qkv  = (const float*)d_in[1];
  const float* b_qkv  = (const float*)d_in[2];
  const float* w_proj = (const float*)d_in[3];
  const float* b_proj = (const float*)d_in[4];
  float* out = (float*)d_out;

  _Float16* ws    = (_Float16*)d_ws;
  _Float16* xh    = ws;
  _Float16* wqkvT = xh    + (size_t)SEQ * HID;        // [2304][768]
  _Float16* wprjT = wqkvT + (size_t)3 * HID * HID;    // [768][768]
  _Float16* Qh    = wprjT + (size_t)HID * HID;        // [12][4096][64]
  _Float16* Kh    = Qh    + (size_t)SEQ * HID;
  _Float16* Vt    = Kh    + (size_t)SEQ * HID;        // [12][64][4096]
  _Float16* Oh    = Vt    + (size_t)SEQ * HID;        // [4096][768]

  // 1) conversions / transposes
  {
    int n = SEQ * HID;
    cvt_f16_kernel<<<(n + 255) / 256, 256, 0, stream>>>(x, xh, n);
  }
  {
    int n = 3 * HID * HID;
    transpose_f16_kernel<<<(n + 255) / 256, 256, 0, stream>>>(w_qkv, wqkvT, HID, 3 * HID);
  }
  {
    int n = HID * HID;
    transpose_f16_kernel<<<(n + 255) / 256, 256, 0, stream>>>(w_proj, wprjT, HID, HID);
  }

  // 2) QKV projection: 128 M-tiles * 36 N-tiles = 4608 waves / 8 = 576 blocks
  gemm_qkv_kernel<<<576, 256, 0, stream>>>(xh, wqkvT, b_qkv, Qh, Kh, Vt);

  // 3) flash attention: 12 heads * 128 q-tiles(32) = 1536 waves / 8 = 192 blocks
  flash_attn_kernel<<<192, 256, 0, stream>>>(Qh, Kh, Vt, Oh);

  // 4) output projection: 128 * 12 = 1536 waves / 8 = 192 blocks
  gemm_proj_kernel<<<192, 256, 0, stream>>>(Oh, wprjT, b_proj, out);
}
